// SpecNormLSTM_5317169512646
// MI455X (gfx1250) — compile-verified, matrix operand order checked by reading, not run on hardware
//
#include <hip/hip_runtime.h>
#include <hip/hip_bf16.h>
#include <math.h>

// ---------------------------------------------------------------------------
// Spectral-norm tanh-RNN for MI455X (gfx1250, wave32, WMMA).
//   T=4096, B=32, I=H=256
// Phase A: sigmas + b_sn (1 block) ; convert w_ih,w_hh -> scaled bf16 in ws
// Phase B: pre = x @ w_ih_sn^T + b_sn  via v_wmma_f32_16x16x32_bf16 -> d_out
// Phase C: sequential scan, single WGP; w_hh held in VGPRs (loop-invariant),
//          h double-buffered in LDS, pre prefetched ahead of the GEMM.
// ---------------------------------------------------------------------------

typedef __attribute__((ext_vector_type(8)))  float   v8f;
typedef __attribute__((ext_vector_type(8)))  __bf16  v8bf;
typedef __attribute__((ext_vector_type(16))) __bf16  v16bf;

#define EPSF 1e-12f
#define TT   4096
#define BB   32
#define HH   256

static __device__ __forceinline__ v16bf cat16(v8bf lo, v8bf hi) {
  return __builtin_shufflevector(lo, hi, 0,1,2,3,4,5,6,7,8,9,10,11,12,13,14,15);
}

static __device__ __forceinline__ v8f wmma_bf16(v16bf a, v16bf b, v8f c) {
  // (neg_a, A, neg_b, B, c_mod, C, reuse_a, reuse_b)
  return __builtin_amdgcn_wmma_f32_16x16x32_bf16(false, a, false, b, (short)0, c,
                                                 false, false);
}

// fast tanh on the recurrence critical path: clamp + v_exp_f32 + v_rcp_f32
static __device__ __forceinline__ float fast_tanh(float x) {
  x = fminf(40.0f, fmaxf(-40.0f, x));
  const float e = __expf(2.0f * x);              // v_exp_f32
  return (e - 1.0f) * __builtin_amdgcn_rcpf(e + 1.0f);
}

// ---------------- Phase A1: sigmas + b_sn ----------------------------------
static __device__ __forceinline__ float reduce256(float v, float* r) {
  const int t = threadIdx.x;
  r[t] = v; __syncthreads();
  #pragma unroll
  for (int s = 128; s > 0; s >>= 1) {
    if (t < s) r[t] += r[t + s];
    __syncthreads();
  }
  float res = r[0]; __syncthreads();
  return res;
}

__global__ __launch_bounds__(256) void k_sigma(
    const float* __restrict__ w_ih, const float* __restrict__ w_hh,
    const float* __restrict__ b_ih, const float* __restrict__ b_hh,
    const float* __restrict__ u_ih, const float* __restrict__ u_hh,
    float* __restrict__ scal /* [0]=inv_sig_ih [1]=inv_sig_hh [2..258)=b_sn */)
{
  __shared__ float red[256];
  __shared__ float u0[256];
  __shared__ float vv[256];
  const int t = threadIdx.x;

  #pragma unroll 1
  for (int which = 0; which < 2; ++which) {
    const float* w = which ? w_hh : w_ih;
    const float* u = which ? u_hh : u_ih;
    // u0 = u / (||u|| + eps)
    float uu = u[t];
    float nu = sqrtf(reduce256(uu * uu, red)) + EPSF;
    u0[t] = uu / nu;
    __syncthreads();
    // v' = W^T u0   (thread t -> column t; coalesced across threads)
    float vp = 0.f;
    #pragma unroll 4
    for (int h = 0; h < HH; ++h) vp += w[h * HH + t] * u0[h];
    float nv = sqrtf(reduce256(vp * vp, red)) + EPSF;
    vv[t] = vp / nv;
    __syncthreads();
    // wv = W v  (thread t -> row t);  sigma = ||wv||^2 / (||wv|| + eps)
    float wv = 0.f;
    #pragma unroll 4
    for (int i = 0; i < HH; ++i) wv += w[t * HH + i] * vv[i];
    float s2 = reduce256(wv * wv, red);
    if (t == 0) {
      float sig = s2 / (sqrtf(s2) + EPSF);
      scal[which] = 1.0f / sig;
    }
    __syncthreads();
  }
  // b_sn = normalize(b_ih) + normalize(b_hh)
  float bi = b_ih[t], bh = b_hh[t];
  float ni = sqrtf(reduce256(bi * bi, red)) + EPSF;
  float nh = sqrtf(reduce256(bh * bh, red)) + EPSF;
  scal[2 + t] = bi / ni + bh / nh;
}

// ---------------- Phase A2: scaled bf16 weight copies ----------------------
__global__ __launch_bounds__(256) void k_cvt(
    const float* __restrict__ w_ih, const float* __restrict__ w_hh,
    const float* __restrict__ scal,
    __bf16* __restrict__ wih_bf, __bf16* __restrict__ whh_bf)
{
  const int i = blockIdx.x * 256 + threadIdx.x;   // 65536 total
  const float si = scal[0], sh = scal[1];
  wih_bf[i] = (__bf16)(w_ih[i] * si);
  whh_bf[i] = (__bf16)(w_hh[i] * sh);
}

// ---------------- Phase B: pre = x @ w_ih_sn^T + b_sn ----------------------
// grid = 1024 blocks x 256 thr (8 waves). Wave -> 16 rows x all 256 cols
// in ONE pass: 16 accumulator tiles (128 VGPRs), x loaded/converted once.
__global__ __launch_bounds__(256) void k_pre(
    const float* __restrict__ x, const __bf16* __restrict__ wih_bf,
    const float* __restrict__ bsn, float* __restrict__ out)
{
  const int wave = threadIdx.x >> 5;
  const int lane = threadIdx.x & 31;
  const int m    = lane & 15;          // A-row / B-col / C-col within tile
  const int lh   = lane >> 4;          // half-wave select
  const int koff = lh * 8;             // K-chunk base offset per ISA layout
  const size_t row0 = (size_t)blockIdx.x * 128 + (size_t)wave * 16;
  const float* xrow = x + (row0 + (size_t)m) * HH;

  v8f acc[16];
  #pragma unroll
  for (int nt = 0; nt < 16; ++nt) acc[nt] = (v8f){0,0,0,0,0,0,0,0};

  #pragma unroll 1
  for (int k0 = 0; k0 < HH; k0 += 32) {
    // A fragment: x rows, convert f32 -> bf16 on the fly
    const float* ap = xrow + k0 + koff;
    v8f a0 = *(const v8f*)ap;
    v8f a1 = *(const v8f*)(ap + 16);
    v16bf a;
    #pragma unroll
    for (int e = 0; e < 8; ++e) { a[e] = (__bf16)a0[e]; a[8+e] = (__bf16)a1[e]; }
    // 16 B fragments (w_ih row n contiguous in k) + 16 WMMA
    #pragma unroll
    for (int nt = 0; nt < 16; ++nt) {
      const __bf16* bp = wih_bf + (size_t)(nt*16 + m) * HH + k0 + koff;
      v16bf b = cat16(*(const v8bf*)bp, *(const v8bf*)(bp + 16));
      acc[nt] = wmma_bf16(a, b, acc[nt]);
    }
  }
  // epilogue: C layout row = i + 8*lh, col = lane&15
  #pragma unroll
  for (int nt = 0; nt < 16; ++nt) {
    const int col = nt*16 + m;
    const float bs = bsn[col];
    #pragma unroll
    for (int i = 0; i < 8; ++i) {
      const size_t row = row0 + (size_t)(i + lh * 8);
      out[row * HH + col] = acc[nt][i] + bs;
    }
  }
}

// ---------------- Phase C: sequential scan ---------------------------------
// Single workgroup, 512 threads = 16 waves. w_hh fragments live in VGPRs
// (invariant over t); LDS holds only the h double-buffer (32 KB).
// Wave -> (mt = wave&1) M-tile, (ng = wave>>1) pair of N-tiles.
__global__ __launch_bounds__(512) void k_scan(
    const float* __restrict__ state, const __bf16* __restrict__ whh_bf,
    float* __restrict__ out)
{
  __shared__ __bf16 hbuf[2][BB * HH];

  const int tid = threadIdx.x;
  const int wave = tid >> 5, lane = tid & 31;
  const int mt = wave & 1, ng = wave >> 1;
  const int m  = lane & 15, lh = lane >> 4;
  const int rowA = mt * 16 + m;
  const int koff = lh * 8;

  // hoist the 16 w_hh B-fragments for this wave's two N-tiles into VGPRs
  v16bf bfrag[2][8];
  #pragma unroll
  for (int nt = 0; nt < 2; ++nt) {
    #pragma unroll
    for (int kk = 0; kk < 8; ++kk) {
      const __bf16* bp = whh_bf + (size_t)(ng*32 + nt*16 + m) * HH + kk*32 + koff;
      bfrag[nt][kk] = cat16(*(const v8bf*)bp, *(const v8bf*)(bp + 16));
    }
  }
  // h0 = state (f32 -> bf16)
  for (int i = tid; i < BB * HH; i += 512)
    hbuf[0][i] = (__bf16)state[i];
  __syncthreads();

  #pragma unroll 1
  for (int t = 0; t < TT; ++t) {
    const __bf16* cur = hbuf[t & 1];
    __bf16*       nxt = hbuf[(t + 1) & 1];

    // prefetch pre[t] (global) so its latency hides under the WMMA chain
    float g[2][8];
    #pragma unroll
    for (int nt = 0; nt < 2; ++nt) {
      const int col = ng*32 + nt*16 + m;
      #pragma unroll
      for (int i = 0; i < 8; ++i) {
        const int row = mt*16 + i + lh*8;
        g[nt][i] = out[((size_t)t * BB + row) * HH + col];
      }
    }

    v8f acc0 = (v8f){0,0,0,0,0,0,0,0};
    v8f acc1 = (v8f){0,0,0,0,0,0,0,0};
    #pragma unroll
    for (int kk = 0; kk < 8; ++kk) {
      const __bf16* ap = cur + rowA * HH + kk*32 + koff;
      v16bf a = cat16(*(const v8bf*)ap, *(const v8bf*)(ap + 16));
      acc0 = wmma_bf16(a, bfrag[0][kk], acc0);
      acc1 = wmma_bf16(a, bfrag[1][kk], acc1);
    }

    // epilogue: tanh(pre + h@W^T), write out (in place over pre) + next-h
    #pragma unroll
    for (int nt = 0; nt < 2; ++nt) {
      const int col = ng*32 + nt*16 + m;
      #pragma unroll
      for (int i = 0; i < 8; ++i) {
        const int row = mt*16 + i + lh*8;
        const size_t idx = ((size_t)t * BB + row) * HH + col;
        const float aval = nt ? acc1[i] : acc0[i];
        const float v = fast_tanh(aval + g[nt][i]);
        out[idx] = v;
        nxt[row * HH + col] = (__bf16)v;
      }
    }
    __syncthreads();
  }

  // h_last: each thread re-copies the elements it wrote at t = TT-1
  #pragma unroll
  for (int nt = 0; nt < 2; ++nt) {
    const int col = ng*32 + nt*16 + m;
    #pragma unroll
    for (int i = 0; i < 8; ++i) {
      const int row = mt*16 + i + lh*8;
      out[(size_t)TT * BB * HH + row * HH + col] =
          out[((size_t)(TT - 1) * BB + row) * HH + col];
    }
  }
}

// ---------------------------------------------------------------------------
extern "C" void kernel_launch(void* const* d_in, const int* in_sizes, int n_in,
                              void* d_out, int out_size, void* d_ws, size_t ws_size,
                              hipStream_t stream) {
  (void)in_sizes; (void)n_in; (void)out_size; (void)ws_size;
  const float* x     = (const float*)d_in[0];
  const float* state = (const float*)d_in[1];
  const float* w_ih  = (const float*)d_in[2];
  const float* w_hh  = (const float*)d_in[3];
  const float* b_ih  = (const float*)d_in[4];
  const float* b_hh  = (const float*)d_in[5];
  const float* u_ih  = (const float*)d_in[6];
  const float* u_hh  = (const float*)d_in[7];
  float* out = (float*)d_out;

  // ws layout: [0..2) inv_sigma, [2..258) b_sn, then bf16 weight copies
  float*  scal   = (float*)d_ws;
  __bf16* wih_bf = (__bf16*)((char*)d_ws + 2048);
  __bf16* whh_bf = wih_bf + HH * HH;

  k_sigma<<<1, 256, 0, stream>>>(w_ih, w_hh, b_ih, b_hh, u_ih, u_hh, scal);
  k_cvt<<<HH * HH / 256, 256, 0, stream>>>(w_ih, w_hh, scal, wih_bf, whh_bf);
  k_pre<<<(TT * BB) / 128, 256, 0, stream>>>(x, wih_bf, scal + 2, out);
  k_scan<<<1, 512, 0, stream>>>(state, whh_bf, out);
}
